// ConvStack_26173530702016
// MI455X (gfx1250) — compile-verified, matrix operand order checked by reading, not run on hardware
//
#include <hip/hip_runtime.h>
#include <hip/hip_bf16.h>
#include <math.h>

typedef __attribute__((ext_vector_type(16))) _Float16 v16h;
typedef __attribute__((ext_vector_type(8)))  _Float16 v8h;
typedef __attribute__((ext_vector_type(8)))  float    v8f;

#define TPB 256

// ---------------- utility kernels ----------------

__global__ void k_fill_f32(float* p, float v, int n) {
  int i = blockIdx.x * blockDim.x + threadIdx.x;
  if (i < n) p[i] = v;
}

__global__ void k_fill_u32(unsigned* p, unsigned v, int n) {
  int i = blockIdx.x * blockDim.x + threadIdx.x;
  if (i < n) p[i] = v;
}

__global__ void k_f32_to_f16(const float* __restrict__ in, _Float16* __restrict__ out, int n) {
  int i = blockIdx.x * blockDim.x + threadIdx.x;
  if (i < n) out[i] = (_Float16)in[i];
}

// W is [K, OC] row-major fp32 -> Wt is [OC, K] row-major fp16
__global__ void k_pack_weight(const float* __restrict__ W, _Float16* __restrict__ Wt, int K, int OC) {
  int i = blockIdx.x * blockDim.x + threadIdx.x;
  if (i >= K * OC) return;
  int k  = i / OC;
  int oc = i - k * OC;
  Wt[(size_t)oc * K + k] = (_Float16)W[i];
}

__global__ void k_build_edges(const int* __restrict__ ei, int E, int N,
                              int* __restrict__ src, int* __restrict__ dst) {
  int i = blockIdx.x * blockDim.x + threadIdx.x;
  if (i < E) { src[i] = ei[i]; dst[i] = ei[E + i]; }
  else if (i < E + N) { src[i] = i - E; dst[i] = i - E; }
}

// ---------------- WMMA GEMM: C[M,NC] = A[M,K](f16) * Bt[NC,K](f16)^T + bias ----------------
// one wave computes a 16x64 output strip (4 accumulators): one A fragment is reused
// across 4 v_wmma_f32_16x16x32_f16 issues per K-step (K stepped by 32).

__global__ void k_gemm_wmma4(const _Float16* __restrict__ A, const _Float16* __restrict__ Bt,
                             const float* __restrict__ bias, float* __restrict__ C,
                             int M, int K, int NC) {
  int gid  = blockIdx.x * blockDim.x + threadIdx.x;
  int wave = gid >> 5;
  int lane = threadIdx.x & 31;
  int nGroups = NC >> 6;                    // 64-wide N groups (OC=256 -> 4, OC=64 -> 1)
  int total   = (M >> 4) * nGroups;
  if (wave >= total) return;                // wave-uniform: EXEC stays all-ones for WMMA
  int mT = wave / nGroups;
  int nG = wave - mT * nGroups;
  int r16  = lane & 15;
  int half = lane >> 4;

  const _Float16* arow = A  + (size_t)(mT * 16 + r16) * K;        // A row (M = lane&15)
  const _Float16* brow = Bt + (size_t)(nG * 64 + r16) * K;        // B col (N = lane&15), tiles at +16K

  v8f acc[4] = {{}, {}, {}, {}};
  union F { v16h v; v8h h[2]; };
  for (int kk = 0; kk < K; kk += 32) {
    F a, b0, b1, b2, b3;
    // 16-bit fragment layout: elems 0..7 -> K = kk + half*8 + j ; elems 8..15 -> K = kk+16+half*8+j
    a.h[0] = *(const v8h*)(arow + kk + half * 8);
    a.h[1] = *(const v8h*)(arow + kk + 16 + half * 8);
    const _Float16* p = brow + kk + half * 8;
    b0.h[0] = *(const v8h*)(p);
    b0.h[1] = *(const v8h*)(p + 16);
    b1.h[0] = *(const v8h*)(p + (size_t)16 * K);
    b1.h[1] = *(const v8h*)(p + (size_t)16 * K + 16);
    b2.h[0] = *(const v8h*)(p + (size_t)32 * K);
    b2.h[1] = *(const v8h*)(p + (size_t)32 * K + 16);
    b3.h[0] = *(const v8h*)(p + (size_t)48 * K);
    b3.h[1] = *(const v8h*)(p + (size_t)48 * K + 16);
    acc[0] = __builtin_amdgcn_wmma_f32_16x16x32_f16(false, a.v, false, b0.v, (short)0, acc[0], false, false);
    acc[1] = __builtin_amdgcn_wmma_f32_16x16x32_f16(false, a.v, false, b1.v, (short)0, acc[1], false, false);
    acc[2] = __builtin_amdgcn_wmma_f32_16x16x32_f16(false, a.v, false, b2.v, (short)0, acc[2], false, false);
    acc[3] = __builtin_amdgcn_wmma_f32_16x16x32_f16(false, a.v, false, b3.v, (short)0, acc[3], false, false);
  }
  // C fragment: VGPR r -> M = r + 8*half, N = lane&15
  int row0 = mT * 16 + half * 8;
#pragma unroll
  for (int s = 0; s < 4; ++s) {
    int col  = nG * 64 + s * 16 + r16;
    float bv = bias[col];
#pragma unroll
    for (int r = 0; r < 8; ++r)
      C[(size_t)(row0 + r) * NC + col] = acc[s][r] + bv;
  }
}

// ---------------- edge-phase kernels (fp32, atomic-based segment softmax) ----------------

__device__ inline float lrelu_s(float v, float s) { return v > 0.f ? v : s * v; }

__device__ inline void atomicMaxF(float* addr, float val) {
  if (val >= 0.f) atomicMax((int*)addr, __float_as_int(val));
  else            atomicMin((unsigned int*)addr, __float_as_uint(val));
}

__global__ void k_edge_score(const float* __restrict__ xl, const float* __restrict__ xr,
                             const int* __restrict__ src, const int* __restrict__ dst,
                             const float* __restrict__ att,
                             float* __restrict__ s, float* __restrict__ m,
                             int E2, int H, int C) {
  int t = blockIdx.x * blockDim.x + threadIdx.x;
  if (t >= E2 * H) return;
  int e = t / H;
  int h = t - e * H;
  int sv = src[e], dv = dst[e];
  const float* pl = xl + (size_t)sv * (H * C) + h * C;
  const float* pr = xr + (size_t)dv * (H * C) + h * C;
  const float* pa = att + h * C;
  float acc = 0.f;
  for (int c = 0; c < C; c += 4) {
    float4 A = *(const float4*)(pl + c);
    float4 B = *(const float4*)(pr + c);
    float4 W = *(const float4*)(pa + c);
    acc += W.x * lrelu_s(A.x + B.x, 0.2f);
    acc += W.y * lrelu_s(A.y + B.y, 0.2f);
    acc += W.z * lrelu_s(A.z + B.z, 0.2f);
    acc += W.w * lrelu_s(A.w + B.w, 0.2f);
  }
  s[t] = acc;
  atomicMaxF(m + (size_t)dv * H + h, acc);
}

__global__ void k_edge_exp(const float* __restrict__ s, const int* __restrict__ dst,
                           const float* __restrict__ m,
                           float* __restrict__ aexp, float* __restrict__ denom,
                           int E2, int H) {
  int t = blockIdx.x * blockDim.x + threadIdx.x;
  if (t >= E2 * H) return;
  int e = t / H;
  int h = t - e * H;
  int dv = dst[e];
  float a = __expf(s[t] - m[(size_t)dv * H + h]);
  aexp[t] = a;
  atomicAdd(denom + (size_t)dv * H + h, a);
}

__global__ void k_edge_agg(const float* __restrict__ aexp, const float* __restrict__ denom,
                           const float* __restrict__ xl,
                           const int* __restrict__ src, const int* __restrict__ dst,
                           float* __restrict__ agg, int E2, int H, int C) {
  long long t = (long long)blockIdx.x * blockDim.x + threadIdx.x;
  int HC = H * C;
  if (t >= (long long)E2 * HC) return;
  int e  = (int)(t / HC);
  int ch = (int)(t - (long long)e * HC);
  int h  = ch / C;
  int dv = dst[e];
  float alpha = aexp[(size_t)e * H + h] / denom[(size_t)dv * H + h];
  atomicAdd(agg + (size_t)dv * HC + ch, alpha * xl[(size_t)src[e] * HC + ch]);
}

__global__ void k_finalize(const float* __restrict__ agg, const float* __restrict__ bias,
                           const float* __restrict__ resid, float* __restrict__ out,
                           int n, int HC) {
  int i = blockIdx.x * blockDim.x + threadIdx.x;
  if (i >= n) return;
  float v = agg[i] + bias[i % HC];
  v = v > 0.f ? v : 0.01f * v;              // outer leaky_relu(0.01)
  if (resid) v += resid[i];                 // layer-3 residual (+x)
  out[i] = v;
}

// ---------------- graph-mode LayerNorm ----------------

__global__ void k_ln_stats(const float* __restrict__ h, const int* __restrict__ batch,
                           float* __restrict__ gsum, float* __restrict__ gsq,
                           int* __restrict__ gcnt, int N, int C) {
  int i = blockIdx.x * blockDim.x + threadIdx.x;
  if (i >= N) return;
  float s = 0.f, q = 0.f;
  const float* p = h + (size_t)i * C;
  for (int c = 0; c < C; c += 4) {
    float4 v = *(const float4*)(p + c);
    s += v.x + v.y + v.z + v.w;
    q += v.x * v.x + v.y * v.y + v.z * v.z + v.w * v.w;
  }
  int g = batch[i];
  atomicAdd(gsum + g, s);
  atomicAdd(gsq + g, q);
  atomicAdd(gcnt + g, 1);
}

__global__ void k_ln_apply(const float* __restrict__ h, const int* __restrict__ batch,
                           const float* __restrict__ gsum, const float* __restrict__ gsq,
                           const int* __restrict__ gcnt,
                           const float* __restrict__ lw, const float* __restrict__ lb,
                           float* __restrict__ out, int N, int C) {
  int t = blockIdx.x * blockDim.x + threadIdx.x;
  if (t >= N * C) return;
  int i = t / C;
  int c = t - i * C;
  int g = batch[i];
  float cnt  = (float)gcnt[g] * (float)C;
  float mean = gsum[g] / cnt;
  float var  = gsq[g] / cnt - mean * mean;
  out[t] = (h[t] - mean) * rsqrtf(var + 1e-5f) * lw[c] + lb[c];
}

// ---------------- host side ----------------

static inline unsigned nblk(long long n) { return (unsigned)((n + TPB - 1) / TPB); }

extern "C" void kernel_launch(void* const* d_in, const int* in_sizes, int n_in,
                              void* d_out, int out_size, void* d_ws, size_t ws_size,
                              hipStream_t stream) {
  const int F = 64, G = 32;
  const float* x   = (const float*)d_in[0];
  const int* ei    = (const int*)d_in[1];
  const int* batch = (const int*)d_in[2];
  int N  = in_sizes[0] / F;       // 20000
  int E  = in_sizes[1] / 2;       // 320000
  int E2 = E + N;                 // with self loops
  const int HCmax = 256;          // max h*F

  // workspace carve-out
  char* w = (char*)d_ws;
  auto carve = [&](size_t bytes) -> void* {
    void* p = (void*)w;
    w += (bytes + 255) & ~(size_t)255;
    return p;
  };
  int*      src2  = (int*)carve((size_t)E2 * 4);
  int*      dst2  = (int*)carve((size_t)E2 * 4);
  _Float16* acth  = (_Float16*)carve((size_t)N * HCmax * 2);
  _Float16* wlT   = (_Float16*)carve((size_t)HCmax * HCmax * 2);
  _Float16* wrT   = (_Float16*)carve((size_t)HCmax * HCmax * 2);
  float*    xl    = (float*)carve((size_t)N * HCmax * 4);
  float*    xr    = (float*)carve((size_t)N * HCmax * 4);
  float*    sbuf  = (float*)carve((size_t)E2 * 4 * 4);
  float*    aexp  = (float*)carve((size_t)E2 * 4 * 4);
  float*    mbuf  = (float*)carve((size_t)N * 4 * 4);
  float*    dnm   = (float*)carve((size_t)N * 4 * 4);
  float*    agg   = (float*)carve((size_t)N * HCmax * 4);
  float*    hA    = (float*)carve((size_t)N * HCmax * 4);
  float*    hB    = (float*)carve((size_t)N * HCmax * 4);
  float*    gsum  = (float*)carve((size_t)G * 4);
  float*    gsq   = (float*)carve((size_t)G * 4);
  int*      gcnt  = (int*)carve((size_t)G * 4);

  // edge list with self loops
  k_build_edges<<<nblk(E2), TPB, 0, stream>>>(ei, E, N, src2, dst2);

  auto run_layer = [&](const float* inAct, int ic, int H,
                       const float* Wl, const float* bl,
                       const float* Wr, const float* br,
                       const float* att, const float* bias,
                       float* outAct, const float* resid) {
    int OC = H * F;   // out channels
    // 1. pack weights (transposed, fp16)
    k_pack_weight<<<nblk((long long)ic * OC), TPB, 0, stream>>>(Wl, wlT, ic, OC);
    k_pack_weight<<<nblk((long long)ic * OC), TPB, 0, stream>>>(Wr, wrT, ic, OC);
    // 2. activation -> fp16
    k_f32_to_f16<<<nblk((long long)N * ic), TPB, 0, stream>>>(inAct, acth, N * ic);
    // 3. WMMA GEMMs (16x64 strip per wave): xl = act@Wl + bl ; xr = act@Wr + br
    long long waves = (long long)(N / 16) * (OC / 64);
    k_gemm_wmma4<<<nblk(waves * 32), TPB, 0, stream>>>(acth, wlT, bl, xl, N, ic, OC);
    k_gemm_wmma4<<<nblk(waves * 32), TPB, 0, stream>>>(acth, wrT, br, xr, N, ic, OC);
    // 4. init softmax scratch + output accumulator
    k_fill_f32<<<nblk((long long)N * H), TPB, 0, stream>>>(mbuf, -INFINITY, N * H);
    k_fill_f32<<<nblk((long long)N * H), TPB, 0, stream>>>(dnm, 0.f, N * H);
    k_fill_f32<<<nblk((long long)N * OC), TPB, 0, stream>>>(agg, 0.f, N * OC);
    // 5-7. attention scores / softmax / aggregation
    k_edge_score<<<nblk((long long)E2 * H), TPB, 0, stream>>>(xl, xr, src2, dst2, att,
                                                              sbuf, mbuf, E2, H, F);
    k_edge_exp<<<nblk((long long)E2 * H), TPB, 0, stream>>>(sbuf, dst2, mbuf, aexp, dnm, E2, H);
    k_edge_agg<<<nblk((long long)E2 * OC), TPB, 0, stream>>>(aexp, dnm, xl, src2, dst2,
                                                             agg, E2, H, F);
    // 8. bias + leaky_relu(0.01) (+ residual for final layer)
    k_finalize<<<nblk((long long)N * OC), TPB, 0, stream>>>(agg, bias, resid, outAct, N * OC, OC);
  };

  auto P = [&](int layer, int j) -> const float* { return (const float*)d_in[3 + 6 * layer + j]; };

  // l0: 64 -> 4x64 ; l1,l2: 256 -> 4x64 ; l3: 256 -> 1x64 (+x residual)
  run_layer(x,  64, 4, P(0,0), P(0,1), P(0,2), P(0,3), P(0,4), P(0,5), hA, nullptr);
  run_layer(hA, 256, 4, P(1,0), P(1,1), P(1,2), P(1,3), P(1,4), P(1,5), hB, nullptr);
  run_layer(hB, 256, 4, P(2,0), P(2,1), P(2,2), P(2,3), P(2,4), P(2,5), hA, nullptr);
  run_layer(hA, 256, 1, P(3,0), P(3,1), P(3,2), P(3,3), P(3,4), P(3,5), hB, x);

  // graph-mode LayerNorm over hB [N, 64]
  const float* lnw = (const float*)d_in[27];
  const float* lnb = (const float*)d_in[28];
  k_fill_f32<<<nblk(G), TPB, 0, stream>>>(gsum, 0.f, G);
  k_fill_f32<<<nblk(G), TPB, 0, stream>>>(gsq, 0.f, G);
  k_fill_u32<<<nblk(G), TPB, 0, stream>>>((unsigned*)gcnt, 0u, G);
  k_ln_stats<<<nblk(N), TPB, 0, stream>>>(hB, batch, gsum, gsq, gcnt, N, F);
  k_ln_apply<<<nblk((long long)N * F), TPB, 0, stream>>>(hB, batch, gsum, gsq, gcnt,
                                                         lnw, lnb, (float*)d_out, N, F);
}
